// MarkovLmEncoderLayer_85426899517711
// MI455X (gfx1250) — matrix operand order repaired and checked by reading, not compiled, not run on hardware
//
#include <hip/hip_runtime.h>
#include <hip/hip_bf16.h>

// ---------------------------------------------------------------------------
// MarkovLmEncoderLayer for MI455X (gfx1250, wave32, WMMA, async LDS copies)
// Pipeline:
//   0) fp32 -> bf16 weight convert + TRANSPOSE (w_up, w_down, w_z) -> [N][K]
//   1) layernorm -> A_up[:, 0:2048] (bf16)
//   2) WMMA GEMM: zp = x_ln @ w_z                          (f32 out)
//   3) z = mu + softplus(ls)*noise -> d_out(z,mu,sigma), A_up[:,2048:], A_down[:,0:64]
//   4) WMMA GEMM: u = [x_ln|z] @ w_up                      (bf16 out)
//   5) rope+pack: q->QB, k->KF(+k_reg,+pad), v->VT transposed (b,h,d,key)
//   6) WMMA flash attention (async double-buffered KV) -> A_down[:, 64:2112]
//   7) silu(gate)*val -> A_down[:, 2112:7744]
//   8) WMMA GEMM + rezero residual: out = hidden + rezero * (A_down @ w_down)
// ---------------------------------------------------------------------------

typedef __bf16 bf16_t;
typedef __attribute__((ext_vector_type(8)))  __bf16 v8bf;
typedef __attribute__((ext_vector_type(16))) __bf16 v16bf;
typedef __attribute__((ext_vector_type(8)))  float  v8f;

// Problem constants
constexpr int B_    = 2;
constexpr int S_    = 2048;
constexpr int HID_  = 2048;
constexpr int H_    = 16;
constexpr int D_    = 128;
constexpr int R_    = 8;
constexpr int LZ_   = 64;
constexpr int MLP_F = 5632;
constexpr int QKV_  = 2048;                   // H*D
constexpr int UCOLS = 3 * QKV_ + 2 * MLP_F;   // 17408
constexpr int AUPC  = HID_ + LZ_;             // 2112
constexpr int ADNC  = LZ_ + QKV_ + MLP_F;     // 7744
constexpr int NT    = B_ * S_;                // 4096 tokens
constexpr int NKEYS = R_ + S_;                // 2056
constexpr int NKP   = 2080;                   // keys padded to 65 * 32
constexpr int NKT   = NKP / 32;               // 65 key tiles

constexpr float EPS_      = 1e-5f;
constexpr float Z_SCALE_  = 0.03125f;               // sqrt(1/1024)
constexpr float SP_SHIFT  = 0.5413248546129181f;    // log(e-1)
constexpr float ATT_SCALE = 0.08838834764831845f;   // 1/sqrt(128)
constexpr float LOG_ROPE  = 9.210340371976184f;     // ln(10000)

// Workspace layout (bytes). Weights stored TRANSPOSED: [N][K], ld = K.
constexpr size_t OFF_WUT = 0;                                        // (UCOLS x AUPC)
constexpr size_t OFF_WDT = OFF_WUT + (size_t)AUPC * UCOLS * 2;       // (HID x ADNC)
constexpr size_t OFF_WZT = OFF_WDT + (size_t)ADNC * HID_ * 2;        // (128 x HID)
constexpr size_t OFF_AUP = OFF_WZT + (size_t)HID_ * 2 * LZ_ * 2;     // A_up bf16
constexpr size_t OFF_ADN = OFF_AUP + (size_t)NT * AUPC * 2;          // A_down bf16
constexpr size_t OFF_U   = OFF_ADN + (size_t)NT * ADNC * 2;          // u bf16
constexpr size_t OFF_ZP  = OFF_U   + (size_t)NT * UCOLS * 2;         // zp f32
constexpr size_t OFF_QB  = OFF_ZP  + (size_t)NT * 2 * LZ_ * 4;       // q (rope) bf16
constexpr size_t OFF_KF  = OFF_QB  + (size_t)NT * QKV_ * 2;          // k_full (b,key,h,d)
constexpr size_t OFF_VT  = OFF_KF  + (size_t)B_ * NKP * QKV_ * 2;    // v_full^T (b,h,d,key)

__device__ __forceinline__ v16bf combine16(v8bf lo, v8bf hi) {
  return __builtin_shufflevector(lo, hi, 0, 1, 2, 3, 4, 5, 6, 7,
                                 8, 9, 10, 11, 12, 13, 14, 15);
}

__device__ __forceinline__ v8f wmma_bf16(v16bf a, v16bf b, v8f c) {
  // D = A(16x32) * B(32x16) + C, f32 accumulate
  return __builtin_amdgcn_wmma_f32_16x16x32_bf16(false, a, false, b,
                                                 (short)0, c, false, false);
}

// CDNA5 async memory->LDS copy (ASYNCcnt-tracked, no VGPR roundtrip).
// LDS operand: generic pointer low 32 bits == LDS byte offset (aperture rule).
__device__ __forceinline__ void async_b128(void* lds, const void* gptr) {
  asm volatile("global_load_async_to_lds_b128 %0, %1, off"
               :: "v"((unsigned)(unsigned long long)lds),
                  "v"((unsigned long long)gptr)
               : "memory");
}
__device__ __forceinline__ void wait_async_le4() {
  asm volatile("s_wait_asynccnt 0x4" ::: "memory");
}
__device__ __forceinline__ void wait_async_le0() {
  asm volatile("s_wait_asynccnt 0x0" ::: "memory");
}

// ---------------------------------------------------------------------------
// 0) fp32 -> bf16 convert + transpose: src[K][N] -> dst[N][K]
//    grid (N/32, K/32), block (32, 8)
// ---------------------------------------------------------------------------
__global__ __launch_bounds__(256) void cvt_t_kernel(
    const float* __restrict__ src, bf16_t* __restrict__ dst, int K, int N) {
  __shared__ bf16_t tile[32][33];
  const int tx = threadIdx.x, ty = threadIdx.y;
  const int n0 = blockIdx.x * 32, k0 = blockIdx.y * 32;
#pragma unroll
  for (int i = 0; i < 4; ++i)
    tile[ty + i * 8][tx] = (bf16_t)src[(size_t)(k0 + ty + i * 8) * N + n0 + tx];
  __syncthreads();
#pragma unroll
  for (int i = 0; i < 4; ++i)
    dst[(size_t)(n0 + ty + i * 8) * K + k0 + tx] = tile[tx][ty + i * 8];
}

// ---------------------------------------------------------------------------
// 1) LayerNorm: one block per token
// ---------------------------------------------------------------------------
__global__ __launch_bounds__(256) void ln_kernel(
    const float* __restrict__ X, bf16_t* __restrict__ Aup) {
  __shared__ float red[256];
  const int tid = threadIdx.x;
  const size_t token = blockIdx.x;
  const float* x = X + token * HID_;

  float v[8], s = 0.f, ss = 0.f;
#pragma unroll
  for (int i = 0; i < 8; ++i) {
    v[i] = x[tid + i * 256];
    s += v[i];
    ss += v[i] * v[i];
  }
  red[tid] = s;
  __syncthreads();
  for (int o = 128; o > 0; o >>= 1) {
    if (tid < o) red[tid] += red[tid + o];
    __syncthreads();
  }
  const float mean = red[0] * (1.f / HID_);
  __syncthreads();
  red[tid] = ss;
  __syncthreads();
  for (int o = 128; o > 0; o >>= 1) {
    if (tid < o) red[tid] += red[tid + o];
    __syncthreads();
  }
  const float var = red[0] * (1.f / HID_) - mean * mean;
  const float rstd = rsqrtf(var + EPS_);

  bf16_t* arow = Aup + token * AUPC;
#pragma unroll
  for (int i = 0; i < 8; ++i)
    arow[tid + i * 256] = (bf16_t)((v[i] - mean) * rstd);
}

// ---------------------------------------------------------------------------
// Generic bf16 WMMA GEMM:  C[M x N] = A[M x K] @ B^T[N x K]
// block = 256 threads = 8 waves; block tile 128x128; wave tile 64x32
// Double-buffered LDS, async global->LDS staging (4 b128 per thread per tile).
// mode 0: bf16 out; mode 1: f32 out; mode 2: f32 out = resid + rezero[col]*acc
// ---------------------------------------------------------------------------
__global__ __launch_bounds__(256) void gemm_bf16_kernel(
    const bf16_t* __restrict__ A, int lda,
    const bf16_t* __restrict__ BT, int ldb,   // BT is [N][K]
    bf16_t* __restrict__ outB, float* __restrict__ outF,
    const float* __restrict__ resid, const float* __restrict__ rezero,
    int ldc, int K, int mode) {
  __shared__ bf16_t As[2][128][32];   // [m][k]
  __shared__ bf16_t Bs[2][128][32];   // [n][k]

  const int tid = threadIdx.x;
  const int lane = tid & 31;
  const int w = tid >> 5;
  const int r = lane & 15;
  const int hh = lane >> 4;
  const int m0 = blockIdx.y * 128;
  const int n0 = blockIdx.x * 128;
  const int wm = (w & 1) * 64;
  const int wn = (w >> 1) * 32;

  v8f zero = {};
  v8f acc[4][2];
#pragma unroll
  for (int j = 0; j < 4; ++j)
#pragma unroll
    for (int i = 0; i < 2; ++i) acc[j][i] = zero;

  // staging: thread covers (row = tid>>1, 16 cols at (tid&1)*16) of each tile
  const int srow = tid >> 1;
  const int skc = (tid & 1) * 16;
  const bf16_t* aptr = A + (size_t)(m0 + srow) * lda + skc;
  const bf16_t* bptr = BT + (size_t)(n0 + srow) * ldb + skc;

  const int nk = K / 32;
  // prologue: stage tile 0
  async_b128(&As[0][srow][skc], aptr);
  async_b128(&As[0][srow][skc + 8], aptr + 8);
  async_b128(&Bs[0][srow][skc], bptr);
  async_b128(&Bs[0][srow][skc + 8], bptr + 8);

  for (int t = 0; t < nk; ++t) {
    const int cur = t & 1;
    if (t + 1 < nk) {
      const int nxt = 1 - cur;
      const size_t ko = (size_t)(t + 1) * 32;
      async_b128(&As[nxt][srow][skc], aptr + ko);
      async_b128(&As[nxt][srow][skc + 8], aptr + ko + 8);
      async_b128(&Bs[nxt][srow][skc], bptr + ko);
      async_b128(&Bs[nxt][srow][skc + 8], bptr + ko + 8);
      wait_async_le4();   // tile t landed; t+1 still in flight
    } else {
      wait_async_le0();
    }
    if (t + 2 < nk) {
      __builtin_prefetch(aptr + (size_t)(t + 2) * 32, 0, 1);
      __builtin_prefetch(bptr + (size_t)(t + 2) * 32, 0, 1);
    }
    __syncthreads();

    v16bf af[4], bfr[2];
#pragma unroll
    for (int j = 0; j < 4; ++j) {
      const bf16_t* base = &As[cur][wm + j * 16 + r][0];
      v8bf lo = *(const v8bf*)(base + hh * 16);
      v8bf hi = *(const v8bf*)(base + hh * 16 + 8);
      af[j] = combine16(lo, hi);
    }
#pragma unroll
    for (int i = 0; i < 2; ++i) {
      const bf16_t* base = &Bs[cur][wn + i * 16 + r][hh * 16];
      v8bf lo = *(const v8bf*)(base);
      v8bf hi = *(const v8bf*)(base + 8);
      bfr[i] = combine16(lo, hi);
    }
#pragma unroll
    for (int j = 0; j < 4; ++j)
#pragma unroll
      for (int i = 0; i < 2; ++i)
        acc[j][i] = wmma_bf16(af[j], bfr[i], acc[j][i]);
    __syncthreads();  // all waves done with buf[cur] before it is re-staged
  }

  // epilogue: C layout -> VGPR v: M = v + 8*half, N = lane(r)
#pragma unroll
  for (int j = 0; j < 4; ++j)
#pragma unroll
    for (int i = 0; i < 2; ++i)
#pragma unroll
      for (int v = 0; v < 8; ++v) {
        const int row = m0 + wm + j * 16 + v + 8 * hh;
        const int col = n0 + wn + i * 16 + r;
        const float val = acc[j][i][v];
        if (mode == 0) {
          outB[(size_t)row * ldc + col] = (bf16_t)val;
        } else if (mode == 1) {
          outF[(size_t)row * ldc + col] = val;
        } else {
          outF[(size_t)row * ldc + col] =
              resid[(size_t)row * ldc + col] + rezero[col] * val;
        }
      }
}

// ---------------------------------------------------------------------------
// 3) z = mu + softplus(ls + shift) * noise
// ---------------------------------------------------------------------------
__global__ __launch_bounds__(256) void z_kernel(
    const float* __restrict__ ZP, const float* __restrict__ noise,
    float* __restrict__ outZ, float* __restrict__ outMu,
    float* __restrict__ outSig, bf16_t* __restrict__ Aup,
    bf16_t* __restrict__ Adn) {
  const size_t idx = (size_t)blockIdx.x * 256 + threadIdx.x;  // NT*LZ total
  const size_t token = idx >> 6;
  const int j = (int)(idx & 63);
  const float mu = Z_SCALE_ * ZP[token * 128 + j];
  const float ls = Z_SCALE_ * ZP[token * 128 + 64 + j] + SP_SHIFT;
  const float sig = (ls > 20.f) ? ls : log1pf(__expf(ls));
  const float z = mu + sig * noise[idx];
  outZ[idx] = z;
  outMu[idx] = mu;
  outSig[idx] = sig;
  Aup[token * AUPC + HID_ + j] = (bf16_t)z;
  Adn[token * ADNC + j] = (bf16_t)z;
}

// ---------------------------------------------------------------------------
// 5) RoPE + pack q/k/v. KF: (b,key,h,d); VT: (b,h,d,key); QB: (b,s,h,d)
// ---------------------------------------------------------------------------
__device__ __forceinline__ float rope_elem(const bf16_t* __restrict__ base,
                                           int d, int s) {
  const float x = (float)base[d];
  if (d >= 64) return x;  // pass-through half
  const int j = d & 31;
  const float inv = __expf(-(float)j * (LOG_ROPE / 32.f));
  const float ang = (float)s * inv;
  float sn, cs;
  __sincosf(ang, &sn, &cs);
  if (d < 32) {
    const float x2 = (float)base[d + 32];
    return x * cs - x2 * sn;
  } else {
    const float x1 = (float)base[d - 32];
    return x1 * sn + x * cs;
  }
}

__global__ __launch_bounds__(256) void rope_pack_kernel(
    const bf16_t* __restrict__ U, const float* __restrict__ k_reg,
    const float* __restrict__ v_reg, bf16_t* __restrict__ QB,
    bf16_t* __restrict__ KF, bf16_t* __restrict__ VT) {
  const int bk = blockIdx.x;  // B * NKP
  const int b = bk / NKP;
  const int key = bk % NKP;
#pragma unroll
  for (int i = 0; i < 8; ++i) {
    const int e = threadIdx.x + i * 256;  // over H*D = 2048
    const int h = e >> 7;
    const int d = e & 127;
    const size_t kdst = (((size_t)b * NKP + key) * H_ + h) * D_ + d;
    const size_t vdst = (((size_t)b * H_ + h) * D_ + d) * NKP + key;
    if (key < R_) {
      KF[kdst] = (bf16_t)k_reg[((size_t)key * H_ + h) * D_ + d];
      VT[vdst] = (bf16_t)v_reg[((size_t)key * H_ + h) * D_ + d];
    } else if (key < NKEYS) {
      const int s = key - R_;
      const bf16_t* urow = U + (size_t)(b * S_ + s) * UCOLS;
      VT[vdst] = urow[2 * QKV_ + h * D_ + d];
      KF[kdst] = (bf16_t)rope_elem(urow + QKV_ + h * D_, d, s);
      QB[((size_t)(b * S_ + s) * H_ + h) * D_ + d] =
          (bf16_t)rope_elem(urow + h * D_, d, s);
    } else {
      KF[kdst] = (bf16_t)0.f;
      VT[vdst] = (bf16_t)0.f;
    }
  }
}

// ---------------------------------------------------------------------------
// 6) Flash attention (WMMA, async double-buffered KV staging)
//    grid = (S/128, B*H). 8 waves x 16 queries.
// ---------------------------------------------------------------------------
__global__ __launch_bounds__(256) void attn_kernel(
    const bf16_t* __restrict__ QB, const bf16_t* __restrict__ KF,
    const bf16_t* __restrict__ VT, const float* __restrict__ mask,
    bf16_t* __restrict__ Adn) {
  __shared__ bf16_t Ks[2][32][128];   // [key][d]
  __shared__ bf16_t Vs[2][128][32];   // [d][key] (already transposed in memory)
  __shared__ bf16_t Ps[8][16][32];    // per-wave P staging

  const int tid = threadIdx.x;
  const int lane = tid & 31;
  const int w = tid >> 5;
  const int r = lane & 15;
  const int hh = lane >> 4;
  const int bh = blockIdx.y;
  const int b = bh >> 4;
  const int head = bh & 15;
  const int q0 = blockIdx.x * 128 + w * 16;

  // Q fragments (16 queries x D=128 -> 4 A fragments), live whole kernel
  v16bf qf[4];
  {
    const bf16_t* qrow = QB + ((size_t)(b * S_ + q0 + r) * H_ + head) * D_;
#pragma unroll
    for (int f = 0; f < 4; ++f) {
      v8bf lo = *(const v8bf*)(qrow + f * 32 + hh * 16);
      v8bf hi = *(const v8bf*)(qrow + f * 32 + hh * 16 + 8);
      qf[f] = combine16(lo, hi);
    }
  }

  v8f zero = {};
  v8f o[8];
#pragma unroll
  for (int i = 0; i < 8; ++i) o[i] = zero;
  float m[8], l[8];
#pragma unroll
  for (int j = 0; j < 8; ++j) {
    m[j] = -3.0e38f;
    l[j] = 0.f;
  }

  // staging geometry
  const int skey = tid >> 3;            // K tile: key row 0..31
  const int sd0 = (tid & 7) * 16;       //         d chunk
  const int vd = tid >> 1;              // V tile: d row 0..127
  const int vk0 = (tid & 1) * 16;       //         key chunk
  const bf16_t* kbase =
      KF + ((size_t)b * NKP * H_ + head) * D_;              // + key*2048 + d
  const bf16_t* vbase =
      VT + (((size_t)b * H_ + head) * D_) * (size_t)NKP;    // + d*NKP + key

  // prologue: stage key-tile 0
  {
    const bf16_t* kp = kbase + (size_t)skey * QKV_ + sd0;
    async_b128(&Ks[0][skey][sd0], kp);
    async_b128(&Ks[0][skey][sd0 + 8], kp + 8);
    const bf16_t* vp = vbase + (size_t)vd * NKP + vk0;
    async_b128(&Vs[0][vd][vk0], vp);
    async_b128(&Vs[0][vd][vk0 + 8], vp + 8);
  }

  for (int t = 0; t < NKT; ++t) {
    const int cur = t & 1;
    const int kk0 = t * 32;
    if (t + 1 < NKT) {
      const int nxt = 1 - cur;
      const bf16_t* kp = kbase + (size_t)(kk0 + 32 + skey) * QKV_ + sd0;
      async_b128(&Ks[nxt][skey][sd0], kp);
      async_b128(&Ks[nxt][skey][sd0 + 8], kp + 8);
      const bf16_t* vp = vbase + (size_t)vd * NKP + kk0 + 32 + vk0;
      async_b128(&Vs[nxt][vd][vk0], vp);
      async_b128(&Vs[nxt][vd][vk0 + 8], vp + 8);
      wait_async_le4();
    } else {
      wait_async_le0();
    }
    __syncthreads();

    // scores: S(16x32) = Q(16x128) @ K^T(128x32) -> 2 C tiles
    v8f s0 = zero, s1 = zero;
#pragma unroll
    for (int f = 0; f < 4; ++f) {
      const bf16_t* kb0 = &Ks[cur][r][f * 32 + hh * 16];
      v16bf b0 = combine16(*(const v8bf*)kb0, *(const v8bf*)(kb0 + 8));
      const bf16_t* kb1 = &Ks[cur][16 + r][f * 32 + hh * 16];
      v16bf b1 = combine16(*(const v8bf*)kb1, *(const v8bf*)(kb1 + 8));
      s0 = wmma_bf16(qf[f], b0, s0);
      s1 = wmma_bf16(qf[f], b1, s1);
    }

    // scale + mask + online softmax
    const int key_a = kk0 + r;
    const int key_b = kk0 + 16 + r;
    float al[8];
#pragma unroll
    for (int j = 0; j < 8; ++j) {
      const int qg = q0 + j + 8 * hh;  // query position in S
      float a = s0[j] * ATT_SCALE;
      float c = s1[j] * ATT_SCALE;
      if (key_a >= R_ && key_a < NKEYS)
        a += mask[(size_t)qg * S_ + (key_a - R_)];
      if (key_b >= R_ && key_b < NKEYS)
        c += mask[(size_t)qg * S_ + (key_b - R_)];
      if (key_a >= NKEYS) a = -1.0e30f;
      if (key_b >= NKEYS) c = -1.0e30f;

      float rm = fmaxf(a, c);
#pragma unroll
      for (int off = 1; off < 16; off <<= 1)
        rm = fmaxf(rm, __shfl_xor(rm, off, 32));
      const float mnew = fmaxf(m[j], rm);
      const float alpha = __expf(m[j] - mnew);
      const float e0 = __expf(a - mnew);
      const float e1 = __expf(c - mnew);
      float rs = e0 + e1;
#pragma unroll
      for (int off = 1; off < 16; off <<= 1) rs += __shfl_xor(rs, off, 32);
      l[j] = l[j] * alpha + rs;
      m[j] = mnew;
      al[j] = alpha;
      // stage P in C layout -> LDS (row = j+8*hh)
      Ps[w][j + 8 * hh][r] = (bf16_t)e0;
      Ps[w][j + 8 * hh][16 + r] = (bf16_t)e1;
    }

    // read P back in A-fragment layout (wave-private region, DS in-order)
    const bf16_t* pp = &Ps[w][r][hh * 16];
    v16bf pf = combine16(*(const v8bf*)pp, *(const v8bf*)(pp + 8));

    // O = diag(alpha) * O + P @ V
#pragma unroll
    for (int i = 0; i < 8; ++i) {
      v8f tacc = o[i];
#pragma unroll
      for (int j = 0; j < 8; ++j) tacc[j] *= al[j];
      const bf16_t* vb = &Vs[cur][i * 16 + r][hh * 16];
      v16bf bv = combine16(*(const v8bf*)vb, *(const v8bf*)(vb + 8));
      o[i] = wmma_bf16(pf, bv, tacc);
    }
    __syncthreads();  // all waves done with buf[cur] before re-staging
  }

  // normalize + write attn_out into A_down[:, LZ + head*128 + d]
  float invl[8];
#pragma unroll
  for (int j = 0; j < 8; ++j) invl[j] = 1.f / l[j];
#pragma unroll
  for (int i = 0; i < 8; ++i)
#pragma unroll
    for (int j = 0; j < 8; ++j) {
      const int q = q0 + j + 8 * hh;
      const int d = i * 16 + r;
      Adn[(size_t)(b * S_ + q) * ADNC + LZ_ + head * D_ + d] =
          (bf16_t)(o[i][j] * invl[j]);
    }
}

// ---------------------------------------------------------------------------
// 7) MLP gate: silu(g) * v -> A_down[:, LZ+QKV : ]
// ---------------------------------------------------------------------------
__global__ __launch_bounds__(256) void mlp_kernel(
    const bf16_t* __restrict__ U, bf16_t* __restrict__ Adn) {
  const size_t idx = (size_t)blockIdx.x * 256 + threadIdx.x;  // NT * MLP_F
  const size_t row = idx / MLP_F;
  const int c = (int)(idx % MLP_F);
  const float g = (float)U[row * UCOLS + 3 * QKV_ + c];
  const float v = (float)U[row * UCOLS + 3 * QKV_ + MLP_F + c];
  const float silu = g * (1.f / (1.f + __expf(-g)));
  Adn[row * ADNC + LZ_ + QKV_ + c] = (bf16_t)(silu * v);
}

// ---------------------------------------------------------------------------
// Launch
// ---------------------------------------------------------------------------
extern "C" void kernel_launch(void* const* d_in, const int* in_sizes, int n_in,
                              void* d_out, int out_size, void* d_ws,
                              size_t ws_size, hipStream_t stream) {
  (void)in_sizes; (void)n_in; (void)out_size; (void)ws_size;

  const float* hidden = (const float*)d_in[0];
  const float* noise  = (const float*)d_in[1];
  const float* mask   = (const float*)d_in[2];
  const float* w_z    = (const float*)d_in[3];
  const float* w_up   = (const float*)d_in[4];
  const float* w_down = (const float*)d_in[5];
  const float* k_reg  = (const float*)d_in[6];
  const float* v_reg  = (const float*)d_in[7];
  const float* rezero = (const float*)d_in[8];

  float* out    = (float*)d_out;
  float* outZ   = out + (size_t)NT * HID_;
  float* outMu  = outZ + (size_t)NT * LZ_;
  float* outSig = outMu + (size_t)NT * LZ_;

  char* ws = (char*)d_ws;
  bf16_t* WUT = (bf16_t*)(ws + OFF_WUT);
  bf16_t* WDT = (bf16_t*)(ws + OFF_WDT);
  bf16_t* WZT = (bf16_t*)(ws + OFF_WZT);
  bf16_t* AUP = (bf16_t*)(ws + OFF_AUP);
  bf16_t* ADN = (bf16_t*)(ws + OFF_ADN);
  bf16_t* U   = (bf16_t*)(ws + OFF_U);
  float*  ZP  = (float*)(ws + OFF_ZP);
  bf16_t* QB  = (bf16_t*)(ws + OFF_QB);
  bf16_t* KF  = (bf16_t*)(ws + OFF_KF);
  bf16_t* VT  = (bf16_t*)(ws + OFF_VT);

  dim3 tb(32, 8);
  // 0) weight convert + transpose ([K][N] -> [N][K])
  cvt_t_kernel<<<dim3(UCOLS / 32, AUPC / 32), tb, 0, stream>>>(w_up, WUT, AUPC,
                                                               UCOLS);
  cvt_t_kernel<<<dim3(HID_ / 32, ADNC / 32), tb, 0, stream>>>(w_down, WDT,
                                                              ADNC, HID_);
  cvt_t_kernel<<<dim3((2 * LZ_) / 32, HID_ / 32), tb, 0, stream>>>(
      w_z, WZT, HID_, 2 * LZ_);

  // 1) layernorm
  ln_kernel<<<NT, 256, 0, stream>>>(hidden, AUP);

  // 2) zp = x_ln @ w_z   (N=128, K=2048)
  gemm_bf16_kernel<<<dim3(1, NT / 128), 256, 0, stream>>>(
      AUP, AUPC, WZT, HID_, nullptr, ZP, nullptr, nullptr, 2 * LZ_, HID_, 1);

  // 3) z / mu / sigma
  z_kernel<<<(NT * LZ_) / 256, 256, 0, stream>>>(ZP, noise, outZ, outMu,
                                                 outSig, AUP, ADN);

  // 4) u = [x_ln|z] @ w_up   (N=17408, K=2112)
  gemm_bf16_kernel<<<dim3(UCOLS / 128, NT / 128), 256, 0, stream>>>(
      AUP, AUPC, WUT, AUPC, U, nullptr, nullptr, nullptr, UCOLS, AUPC, 0);

  // 5) rope + pack q/k/v
  rope_pack_kernel<<<B_ * NKP, 256, 0, stream>>>(U, k_reg, v_reg, QB, KF, VT);

  // 6) flash attention
  attn_kernel<<<dim3(S_ / 128, B_ * H_), 256, 0, stream>>>(QB, KF, VT, mask,
                                                           ADN);

  // 7) mlp elementwise
  mlp_kernel<<<(NT * MLP_F) / 256, 256, 0, stream>>>(U, ADN);

  // 8) out = hidden + rezero * (A_down @ w_down)   (N=2048, K=7744)
  gemm_bf16_kernel<<<dim3(HID_ / 128, NT / 128), 256, 0, stream>>>(
      ADN, ADNC, WDT, ADNC, nullptr, out, hidden, rezero, HID_, ADNC, 2);
}
